// plain_GNN_5858335391829
// MI455X (gfx1250) — matrix operand order repaired and checked by reading, not compiled, not run on hardware
//
#include <hip/hip_runtime.h>
#include <hip/hip_bf16.h>

typedef __bf16 bf16_t;
typedef __attribute__((ext_vector_type(16))) __bf16 v16bf;
typedef __attribute__((ext_vector_type(8)))  __bf16 v8bf;
typedef __attribute__((ext_vector_type(8)))  float  v8f;

#define NNODES 20000
#define NEDGES 160000
#define NF 512
#define NH 512
#define NC 64
#define NLAY 4

#define GEMM_K 512
#define ROWS_PER_WAVE 32
#define WAVES_PER_BLK 4
#define BLOCK_ROWS (ROWS_PER_WAVE * WAVES_PER_BLK)   // 128
#define BPAD 8
#define BSTRIDE (GEMM_K + BPAD)                       // 520 elems -> 1040B row stride

// ---------------- degree / normalization ----------------
__global__ void k_count_deg(const int* __restrict__ dst, float* __restrict__ deg, int E) {
    int i = blockIdx.x * blockDim.x + threadIdx.x;
    if (i < E) atomicAdd(&deg[dst[i]], 1.0f);
}

__global__ void k_finalize_dinv(float* __restrict__ deg, int n) {
    int i = blockIdx.x * blockDim.x + threadIdx.x;
    if (i < n) deg[i] = rsqrtf(deg[i] + 1.0f);   // +1 = self loop
}

// ---------------- conversions ----------------
__global__ void k_f32_to_bf16(const float* __restrict__ in, bf16_t* __restrict__ out, int n) {
    int i = blockIdx.x * blockDim.x + threadIdx.x;
    if (i < n) out[i] = (bf16_t)in[i];
}

// Wt[n][k] = bf16(W[k][n]);  W is K x Ncols row-major
__global__ void k_transpose_bf16(const float* __restrict__ W, bf16_t* __restrict__ Wt,
                                 int K, int Ncols) {
    int i = blockIdx.x * blockDim.x + threadIdx.x;
    if (i < K * Ncols) {
        int k = i / Ncols, n = i % Ncols;
        Wt[(size_t)n * K + k] = (bf16_t)W[i];
    }
}

// ---------------- WMMA GEMM with LDS-staged weights ----------------
// out[M x Nout] = act(A[M x K=512] @ Wt^T + bias).  Wt is [Nout x 512] bf16 (pre-transposed).
// Block: 4 waves, 128x64 output tile. B strip (64x512 bf16) staged once in LDS (padded rows),
// shared by all waves. Each wave computes 32x64 (2 row-groups x 4 col-tiles, 8 WMMAs/K-step).
// grid = (ceil(M/128), Nout/64), block = 128 threads.
__global__ __launch_bounds__(128)
void k_wmma_gemm(const bf16_t* __restrict__ A, const bf16_t* __restrict__ Wt,
                 const float* __restrict__ bias, float* __restrict__ outF,
                 bf16_t* __restrict__ outH, int M, int Nout, int doRelu) {
    __shared__ bf16_t Bs[64 * BSTRIDE];              // 66.5 KB

    const int tid  = threadIdx.x;
    const int wave = tid >> 5;
    const int lane = tid & 31;
    const int r    = lane & 15;
    const int half = lane >> 4;
    const int n0   = blockIdx.y << 6;

    // stage B strip: 64 contiguous Wt rows starting at n0 (64*512 bf16), pad to BSTRIDE
    {
        const float4* gsrc = (const float4*)(Wt + (size_t)n0 * GEMM_K);
        for (int idx = tid; idx < 64 * (GEMM_K / 8); idx += blockDim.x) {
            const int row = idx >> 6;                // 64 float4-chunks per row
            const int ch  = idx & 63;
            *(float4*)(Bs + row * BSTRIDE + ch * 8) = gsrc[row * 64 + ch];
        }
    }
    __syncthreads();

    const int mw = blockIdx.x * BLOCK_ROWS + wave * ROWS_PER_WAVE;

    // clamped A rows (EXEC stays all-ones for WMMA); stores guarded below
    int row0 = mw + r;      if (row0 > M - 1) row0 = M - 1;
    int row1 = mw + 16 + r; if (row1 > M - 1) row1 = M - 1;
    const bf16_t* arow0 = A + (size_t)row0 * GEMM_K;
    const bf16_t* arow1 = A + (size_t)row1 * GEMM_K;

    v8f c[2][4] = {};

    for (int k0 = 0; k0 < GEMM_K; k0 += 32) {
        v8bf a0lo = *(const v8bf*)(arow0 + k0 + 8 * half);
        v8bf a0hi = *(const v8bf*)(arow0 + k0 + 16 + 8 * half);
        v8bf a1lo = *(const v8bf*)(arow1 + k0 + 8 * half);
        v8bf a1hi = *(const v8bf*)(arow1 + k0 + 16 + 8 * half);
        v16bf a0 = __builtin_shufflevector(a0lo, a0hi,
                     0,1,2,3,4,5,6,7,8,9,10,11,12,13,14,15);
        v16bf a1 = __builtin_shufflevector(a1lo, a1hi,
                     0,1,2,3,4,5,6,7,8,9,10,11,12,13,14,15);
        if (k0 + 32 < GEMM_K) __builtin_prefetch(arow0 + k0 + 64, 0, 1);
#pragma unroll
        for (int j = 0; j < 4; j++) {
            const bf16_t* bp = Bs + (j * 16 + r) * BSTRIDE + k0 + 8 * half;
            v8bf blo = *(const v8bf*)bp;
            v8bf bhi = *(const v8bf*)(bp + 16);
            v16bf b = __builtin_shufflevector(blo, bhi,
                        0,1,2,3,4,5,6,7,8,9,10,11,12,13,14,15);
            c[0][j] = __builtin_amdgcn_wmma_f32_16x16x32_bf16(
                          false, a0, false, b, (short)0, c[0][j], false, false);
            c[1][j] = __builtin_amdgcn_wmma_f32_16x16x32_bf16(
                          false, a1, false, b, (short)0, c[1][j], false, false);
        }
    }

    // epilogue: C/D layout -> row = base + 8*half + v, col = n0 + 16*j + r
#pragma unroll
    for (int g = 0; g < 2; g++) {
        const int rowb = mw + g * 16 + 8 * half;
#pragma unroll
        for (int j = 0; j < 4; j++) {
            const int n  = n0 + 16 * j + r;
            const float bv = bias ? bias[n] : 0.0f;
#pragma unroll
            for (int v = 0; v < 8; v++) {
                const int row = rowb + v;
                if (row < M) {
                    float val = c[g][j][v] + bv;
                    if (doRelu) val = fmaxf(val, 0.0f);
                    size_t idx = (size_t)row * Nout + n;
                    if (outF) outF[idx] = val;
                    else      outH[idx] = (bf16_t)val;
                }
            }
        }
    }
}

// ---------------- edge scatter (message passing) ----------------
// one block per edge; 128 threads x float4 = 512 feats; atomics resolve in L2
__global__ __launch_bounds__(128)
void k_scatter(const float* __restrict__ hW, const int* __restrict__ src,
               const int* __restrict__ dst, const float* __restrict__ dinv,
               float* __restrict__ agg) {
    const int e = blockIdx.x;
    const int s = src[e], d = dst[e];
    const float nrm = dinv[s] * dinv[d];
    const float4 v = ((const float4*)(hW + (size_t)s * NH))[threadIdx.x];
    float* o = agg + (size_t)d * NH + threadIdx.x * 4;
    atomicAdd(o + 0, v.x * nrm);
    atomicAdd(o + 1, v.y * nrm);
    atomicAdd(o + 2, v.z * nrm);
    atomicAdd(o + 3, v.w * nrm);
}

// ---------------- self-loop + bias + relu + bf16 convert ----------------
__global__ __launch_bounds__(128)
void k_selfloop_bias_relu(const float* __restrict__ hW, const float* __restrict__ agg,
                          const float* __restrict__ dinv, const float* __restrict__ bias,
                          bf16_t* __restrict__ hout) {
    const int vtx = blockIdx.x;
    const int t   = threadIdx.x;
    const float sl = dinv[vtx] * dinv[vtx];
    const float4 hw = ((const float4*)(hW  + (size_t)vtx * NH))[t];
    const float4 ag = ((const float4*)(agg + (size_t)vtx * NH))[t];
    const float4 b  = ((const float4*)bias)[t];
    bf16_t* o = hout + (size_t)vtx * NH + t * 4;
    o[0] = (bf16_t)fmaxf(ag.x + hw.x * sl + b.x, 0.0f);
    o[1] = (bf16_t)fmaxf(ag.y + hw.y * sl + b.y, 0.0f);
    o[2] = (bf16_t)fmaxf(ag.z + hw.z * sl + b.z, 0.0f);
    o[3] = (bf16_t)fmaxf(ag.w + hw.w * sl + b.w, 0.0f);
}

// ---------------- host launcher ----------------
extern "C" void kernel_launch(void* const* d_in, const int* in_sizes, int n_in,
                              void* d_out, int out_size, void* d_ws, size_t ws_size,
                              hipStream_t stream) {
    const float* x       = (const float*)d_in[0];
    const int*   ei      = (const int*)  d_in[1];
    const float* W_enc   = (const float*)d_in[2];
    const float* b_enc   = (const float*)d_in[3];
    const float* W_convs = (const float*)d_in[4];
    const float* b_convs = (const float*)d_in[5];
    const float* W_dec   = (const float*)d_in[6];
    const float* b_dec   = (const float*)d_in[7];
    float*       out     = (float*)d_out;

    const int* src = ei;
    const int* dst = ei + NEDGES;

    // workspace carve-out (256B aligned chunks)
    char* ws = (char*)d_ws;
    size_t o = 0;
    auto alloc = [&](size_t bytes) -> char* {
        char* p = ws + o;
        o += (bytes + 255) & ~(size_t)255;
        return p;
    };
    float*  dinv = (float*) alloc((size_t)NNODES * 4);
    bf16_t* xb   = (bf16_t*)alloc((size_t)NNODES * NF * 2);
    bf16_t* hb   = (bf16_t*)alloc((size_t)NNODES * NH * 2);
    float*  hW   = (float*) alloc((size_t)NNODES * NH * 4);
    float*  agg  = (float*) alloc((size_t)NNODES * NH * 4);
    bf16_t* wtE  = (bf16_t*)alloc((size_t)NF * NH * 2);
    bf16_t* wtC  = (bf16_t*)alloc((size_t)NLAY * NH * NH * 2);
    bf16_t* wtD  = (bf16_t*)alloc((size_t)NH * NC * 2);

    // 1) degree -> dinv = rsqrt(deg+1)
    hipMemsetAsync(dinv, 0, (size_t)NNODES * 4, stream);
    k_count_deg<<<(NEDGES + 255) / 256, 256, 0, stream>>>(dst, dinv, NEDGES);
    k_finalize_dinv<<<(NNODES + 255) / 256, 256, 0, stream>>>(dinv, NNODES);

    // 2) convert inputs/weights to bf16 (weights transposed to [N][K])
    k_f32_to_bf16<<<((size_t)NNODES * NF + 255) / 256, 256, 0, stream>>>(x, xb, NNODES * NF);
    k_transpose_bf16<<<(NF * NH + 255) / 256, 256, 0, stream>>>(W_enc, wtE, NF, NH);
    for (int l = 0; l < NLAY; l++)
        k_transpose_bf16<<<(NH * NH + 255) / 256, 256, 0, stream>>>(
            W_convs + (size_t)l * NH * NH, wtC + (size_t)l * NH * NH, NH, NH);
    k_transpose_bf16<<<(NH * NC + 255) / 256, 256, 0, stream>>>(W_dec, wtD, NH, NC);

    const int gx = (NNODES + BLOCK_ROWS - 1) / BLOCK_ROWS;   // 157

    // 3) encoder: hb = bf16(relu(x @ W_enc + b_enc))
    dim3 gE(gx, NH / 64);
    k_wmma_gemm<<<gE, 128, 0, stream>>>(xb, wtE, b_enc, nullptr, hb, NNODES, NH, 1);

    // 4) GCN layers
    for (int l = 0; l < NLAY; l++) {
        k_wmma_gemm<<<gE, 128, 0, stream>>>(hb, wtC + (size_t)l * NH * NH,
                                            nullptr, hW, nullptr, NNODES, NH, 0);
        hipMemsetAsync(agg, 0, (size_t)NNODES * NH * 4, stream);
        k_scatter<<<NEDGES, 128, 0, stream>>>(hW, src, dst, dinv, agg);
        k_selfloop_bias_relu<<<NNODES, 128, 0, stream>>>(hW, agg, dinv,
                                                         b_convs + (size_t)l * NH, hb);
    }

    // 5) decoder: out = hb @ W_dec + b_dec (f32)
    dim3 gD(gx, NC / 64);
    k_wmma_gemm<<<gD, 128, 0, stream>>>(hb, wtD, b_dec, out, nullptr, NNODES, NC, 0);
}
